// QNN_7559142440985
// MI455X (gfx1250) — matrix operand order repaired and checked by reading, not compiled, not run on hardware
//
#include <hip/hip_runtime.h>

// ---------------------------------------------------------------------------
// Fused 12-qubit QNN forward for MI455X (gfx1250, wave32, WMMA).
//
// psi stays REAL (real init, real Ry, CNOT=permutation) -> bf16 state in LDS.
// Per layer: 3 nibble contractions (16x16 Kronecker G matrices) via
// v_wmma_f32_16x16x32_bf16 (batch rows = M, K=16 zero-padded to 32),
// CNOT-chain permutation s -> s^(s>>1) folded into pass-2 store addresses.
// Passes are template-specialized (compile-time addressing, no exec-mask
// branching); block loop has a constant trip count and is unrolled x2 so two
// independent WMMAs are in flight (fills the WMMA->VALU hazard slots).
// ---------------------------------------------------------------------------

typedef __bf16 bf16;
typedef __attribute__((ext_vector_type(16))) __bf16 v16bf;
typedef __attribute__((ext_vector_type(8)))  __bf16 v8bf;
typedef __attribute__((ext_vector_type(8)))  float  v8f;

#define QN_LAYERS   8
#define QN_NQ       12
#define QN_NCLASS   10
#define QN_BATCH    2048
#define QN_FEAT     3072
#define QN_DIM      4096
#define QN_TILE     16          // batch rows per workgroup (= WMMA M)
#define QN_RS       4104        // bf16 row stride: 8208B = 16B-aligned, 4-bank skew
#define QN_NTHR     256         // 8 wave32s
#define QN_NWAVE    8
#define QN_BLKS     (256 / QN_NWAVE)   // blocks per wave per pass (32)

// dynamic LDS layout (bytes)
#define LDS_BUF     (QN_TILE * QN_RS * 2)                 // one bf16 state buffer
#define LDS_TOTAL   (2*LDS_BUF + QN_LAYERS*QN_NQ*4 + 3*256*2 + QN_TILE*4 + QN_TILE*QN_NCLASS*4)

// One contraction pass, fully specialized on PASS (0: axis c, 1: axis b, 2: axis a).
template <int PASS>
__device__ __forceinline__ void contract_pass(const bf16* __restrict__ src,
                                              bf16* __restrict__ dst,
                                              const bf16* __restrict__ s_G,
                                              int lm, int lhi, int wave) {
    // B fragment: 32x16 bf16, K rows 16..31 zero (K padding).
    // lane L holds B[k][N=L%16] = G[L%16][k], k = 8*(L/16)+j for j<8.
    v16bf bfrag;
    #pragma unroll
    for (int j = 0; j < 16; ++j) bfrag[j] = (bf16)0.f;
    {
        const bf16* grow = s_G + (2 - PASS) * 256 + lm * 16 + lhi * 8;
        #pragma unroll
        for (int j = 0; j < 8; ++j) bfrag[j] = grow[j];
    }
    const v8f czero = {0.f, 0.f, 0.f, 0.f, 0.f, 0.f, 0.f, 0.f};

    const bf16* arow = src + lm * QN_RS;    // A row for this lane (M = lm)
    const int   k0   = lhi * 8;             // K half handled by this half-wave

    // Constant trip count -> scalar loop control; x2 unroll -> two WMMAs in
    // flight per iteration to cover the WMMA->VALU data-hazard slots.
    #pragma unroll 2
    for (int it = 0; it < QN_BLKS; ++it) {
        const int blk = wave + it * QN_NWAVE;
        const int u   = blk >> 4;
        const int v_  = blk & 15;

        // A fragment: M=lm, K=(lhi ? 8..15 : 0..7); K 16..31 = 0.
        v16bf afrag;
        #pragma unroll
        for (int j = 8; j < 16; ++j) afrag[j] = (bf16)0.f;

        if (PASS == 0) {
            // contiguous axis: one 16B LDS load (ds_load_b128)
            const v8bf a8 = *(const v8bf*)(arow + blk * 16 + k0);
            #pragma unroll
            for (int j = 0; j < 8; ++j) afrag[j] = a8[j];
        } else {
            #pragma unroll
            for (int j = 0; j < 8; ++j) {
                const int k   = k0 + j;
                const int col = (PASS == 1) ? (u * 256 + k * 16 + v_)   // contract b
                                            : (k * 256 + u * 16 + v_); // contract a
                afrag[j] = arow[col];
            }
        }

        const v8f d = __builtin_amdgcn_wmma_f32_16x16x32_bf16(
                          false, afrag, false, bfrag, (short)0, czero, false, false);

        // Store D; PASS 2 folds the CNOT-chain permutation s -> s^(s>>1).
        #pragma unroll
        for (int r = 0; r < 8; ++r) {
            const int mo = r + lhi * 8;
            int col;
            if (PASS == 0)      col = blk * 16 + lm;
            else if (PASS == 1) col = u * 256 + lm * 16 + v_;
            else { const int s2 = lm * 256 + u * 16 + v_; col = s2 ^ (s2 >> 1); }
            dst[mo * QN_RS + col] = (bf16)d[r];
        }
    }
}

__global__ void __launch_bounds__(QN_NTHR)
qnn_fused_kernel(const float* __restrict__ x,
                 const float* __restrict__ angles,
                 const float* __restrict__ W,
                 const float* __restrict__ bias,
                 float* __restrict__ out) {
    extern __shared__ char smem_raw[];
    bf16*  buf0    = (bf16*)smem_raw;                       // [16][QN_RS]
    bf16*  buf1    = buf0 + QN_TILE * QN_RS;                // [16][QN_RS]
    float* s_ang   = (float*)(buf1 + QN_TILE * QN_RS);      // [8][12]
    bf16*  s_G     = (bf16*)(s_ang + QN_LAYERS * QN_NQ);    // [3][16][16]
    float* s_sumsq = (float*)(s_G + 3 * 256);               // [16]
    float* s_acc   = s_sumsq + QN_TILE;                     // [16][10]

    const int tid  = threadIdx.x;
    const int tile = blockIdx.x;
    const int m    = tid >> 4;          // row 0..15 (for strided phases)
    const int l16  = tid & 15;

    const int lane = tid & 31;
    const int lm   = lane & 15;         // fragment row/col index
    const int lhi  = lane >> 4;         // half-wave select
    const int wave = tid >> 5;          // 0..7

    // ---- init: angles to LDS, zero reductions ----
    if (tid < QN_LAYERS * QN_NQ) s_ang[tid] = angles[tid];
    if (tid < QN_TILE)           s_sumsq[tid] = 0.f;
    if (tid < QN_TILE * QN_NCLASS) s_acc[tid] = 0.f;
    __syncthreads();

    // ---- L2 row norms (x read #1, coalesced) ----
    {
        const float* xrow = x + (size_t)(tile * QN_TILE + m) * QN_FEAT;
        float acc = 0.f;
        for (int k = l16; k < QN_FEAT; k += 16) { float v = xrow[k]; acc += v * v; }
        atomicAdd(&s_sumsq[m], acc);     // ds_add_f32
    }
    __syncthreads();

    // ---- normalized initial state -> buf0 (bf16), zero-pad FEAT..DIM ----
    {
        const float* xrow = x + (size_t)(tile * QN_TILE + m) * QN_FEAT;
        const float  rn   = rsqrtf(s_sumsq[m]);
        bf16* row = buf0 + m * QN_RS;
        for (int k = l16; k < QN_DIM; k += 16)
            row[k] = (bf16)(k < QN_FEAT ? xrow[k] * rn : 0.f);
    }
    __syncthreads();

    bf16* src = buf0;
    bf16* dst = buf1;

    // ---- 8 circuit layers ----
    for (int l = 0; l < QN_LAYERS; ++l) {
        // Build the three 16x16 Kronecker rotation matrices for this layer.
        // Nibble g covers qubits 4g..4g+3; qubit 4g+p maps to nibble bit (3-p).
        for (int t = tid; t < 3 * 256; t += QN_NTHR) {
            int g = t >> 8, ij = t & 255, i = ij >> 4, j = ij & 15;
            float v = 1.f;
            #pragma unroll
            for (int p = 0; p < 4; ++p) {
                float a = s_ang[l * QN_NQ + 4 * g + p];
                float c = cosf(a), s = sinf(a);
                int bi = (i >> (3 - p)) & 1, bj = (j >> (3 - p)) & 1;
                v *= bi ? (bj ? c : s) : (bj ? -s : c);
            }
            s_G[t] = (bf16)v;
        }
        __syncthreads();

        contract_pass<0>(src, dst, s_G, lm, lhi, wave);
        __syncthreads();
        { bf16* t2 = src; src = dst; dst = t2; }

        contract_pass<1>(src, dst, s_G, lm, lhi, wave);
        __syncthreads();
        { bf16* t2 = src; src = dst; dst = t2; }

        contract_pass<2>(src, dst, s_G, lm, lhi, wave);
        __syncthreads();
        { bf16* t2 = src; src = dst; dst = t2; }
    }
    // 24 swaps total -> final state back in buf0 (== src)

    // ---- readout: logits = |psi|^2 @ W^T + b  (W stays hot in L2) ----
    {
        const bf16* row = src + m * QN_RS;
        float acc[QN_NCLASS];
        #pragma unroll
        for (int c = 0; c < QN_NCLASS; ++c) acc[c] = 0.f;
        for (int k = l16; k < QN_DIM; k += 16) {
            float p = (float)row[k];
            p = p * p;
            #pragma unroll
            for (int c = 0; c < QN_NCLASS; ++c)
                acc[c] += p * W[c * QN_DIM + k];
        }
        #pragma unroll
        for (int c = 0; c < QN_NCLASS; ++c)
            atomicAdd(&s_acc[m * QN_NCLASS + c], acc[c]);
    }
    __syncthreads();
    if (tid < QN_TILE * QN_NCLASS) {
        const int mm = tid / QN_NCLASS, cc = tid % QN_NCLASS;
        out[(size_t)(tile * QN_TILE + mm) * QN_NCLASS + cc] = s_acc[tid] + bias[cc];
    }
}

extern "C" void kernel_launch(void* const* d_in, const int* in_sizes, int n_in,
                              void* d_out, int out_size, void* d_ws, size_t ws_size,
                              hipStream_t stream) {
    (void)in_sizes; (void)n_in; (void)d_ws; (void)ws_size; (void)out_size;
    const float* x      = (const float*)d_in[0];
    const float* angles = (const float*)d_in[1];
    const float* W      = (const float*)d_in[2];
    const float* bias   = (const float*)d_in[3];
    float*       out    = (float*)d_out;

    // Opt in to >64KB dynamic LDS (no-op / ignored error if not required).
    (void)hipFuncSetAttribute((const void*)qnn_fused_kernel,
                              hipFuncAttributeMaxDynamicSharedMemorySize,
                              (int)LDS_TOTAL);

    dim3 grid(QN_BATCH / QN_TILE);   // 128 workgroups, one 16-row tile each
    dim3 block(QN_NTHR);             // 8 wave32s
    qnn_fused_kernel<<<grid, block, LDS_TOTAL, stream>>>(x, angles, W, bias, out);
}
